// TransformerVAE_36636071035091
// MI455X (gfx1250) — matrix operand order repaired and checked by reading, not compiled, not run on hardware
//
#include <hip/hip_runtime.h>
#include <math.h>
#include <stdint.h>

// ---------------------------------------------------------------------------
// Transformer-VAE forward for MI455X (gfx1250, wave32).
//  * all GEMMs via v_wmma_f32_16x16x32_f16 (f32 accumulate)
//  * activations + weights held in f16; weights pre-transposed so every
//    LDS tile is staged with global_load_async_to_lds_b128 (ASYNCcnt),
//    double-buffered so copies overlap WMMA.
// ---------------------------------------------------------------------------

typedef __attribute__((ext_vector_type(16))) _Float16 v16h;
typedef __attribute__((ext_vector_type(8)))  float    v8f;

#define B_DIM   64
#define S_DIM   512
#define D_DIM   512
#define DK      64
#define DFF_DIM 2048
#define LAT_DIM 128
#define INP_DIM 64
#define SB      (S_DIM * B_DIM)   // 32768 token rows, (s,b) order
#define FF_CHUNK 4096

// ---- WMMA fragment loaders (CDNA5 ISA 7.12.2 layouts) ----------------------
__device__ __forceinline__ v16h frag_a(const _Float16* rowp, int k0, int lane) {
  const int kb = k0 + ((lane >> 4) << 3);
  v16h f;
#pragma unroll
  for (int v = 0; v < 4; ++v) {
    f[2 * v]     = rowp[kb + 2 * v];
    f[2 * v + 1] = rowp[kb + 2 * v + 1];
    f[8 + 2 * v] = rowp[kb + 16 + 2 * v];
    f[9 + 2 * v] = rowp[kb + 16 + 2 * v + 1];
  }
  return f;
}

__device__ __forceinline__ v16h frag_b(const _Float16* rowp, int k0, int lane) {
  const int kb = k0 + ((lane >> 4) << 3);
  v16h f;
#pragma unroll
  for (int v = 0; v < 4; ++v) {
    f[2 * v]     = rowp[kb + 2 * v];
    f[2 * v + 1] = rowp[kb + 2 * v + 1];
    f[8 + 2 * v] = rowp[kb + 16 + 2 * v];
    f[9 + 2 * v] = rowp[kb + 16 + 2 * v + 1];
  }
  return f;
}

__device__ __forceinline__ v8f wmma16(v16h a, v16h b, v8f c) {
  return __builtin_amdgcn_wmma_f32_16x16x32_f16(false, a, false, b, (short)0, c,
                                                false, false);
}

// 16B global -> LDS async copy (ASYNCcnt-tracked). lds_off = byte offset in LDS.
__device__ __forceinline__ void async_b128(uint32_t lds_off, const void* gaddr) {
  asm volatile("global_load_async_to_lds_b128 %0, %1, off"
               :: "v"(lds_off), "v"(gaddr) : "memory");
}

// ---------------------------------------------------------------------------
// GEMM: C[M,N] = epi(A[M,K] * Bt[N,K]^T), A/Bt f16, acc f32.
//  AMAP: 0 identity | 1 x-permute ((r%64)*512 + r/64) | 2 z-broadcast (r%64)
//  CMAP: 0 identity | 1 recon permute ((m%64)*512 + m/64)
//  Tile BM x 64, K-step 32, LDS rows padded to 80B (16B aligned, 20-bank stride)
// ---------------------------------------------------------------------------
template <int BM, int AMAP, int CMAP, int PE, int GELU, typename CT>
__global__ __launch_bounds__(BM * 2)
void gemm_wmma(const _Float16* __restrict__ A, int lda,
               const _Float16* __restrict__ Bt,   // [N][K], pre-transposed
               CT* __restrict__ C, int ldc, int K,
               const float* __restrict__ bias,
               const _Float16* __restrict__ res, float alpha) {
  constexpr int NT = BM * 2;           // threads (BM/16 waves * 32)
  constexpr int AH = 40;               // halves per LDS row (80 bytes)
  constexpr int BUF_H = BM * AH + 64 * AH;
  __shared__ _Float16 smem[2 * BUF_H];

  const int tid  = threadIdx.x;
  const int wave = tid >> 5;
  const int lane = tid & 31;
  const int i16  = lane & 15;
  const int bm   = blockIdx.y * BM;
  const int bn   = blockIdx.x * 64;
  const uint32_t lds0 = (uint32_t)(uintptr_t)&smem[0];

  v8f acc[4] = {};
  const int nk = K >> 5;

  auto stage = [&](int kt, int buf) {
    const int k0 = kt << 5;
    const uint32_t ab = lds0 + (uint32_t)buf * (BUF_H * 2);
    const uint32_t bb = ab + BM * 80;
#pragma unroll
    for (int j = 0; j < (BM * 4) / NT; ++j) {       // A: BM rows x 4 chunks
      const int idx = tid + j * NT;
      const int row = idx >> 2, ch = idx & 3;
      const int gr  = bm + row;
      long arow = gr;
      if constexpr (AMAP == 1) arow = (long)(gr & 63) * S_DIM + (gr >> 6);
      if constexpr (AMAP == 2) arow = gr & 63;
      async_b128(ab + row * 80 + ch * 16, A + arow * (long)lda + k0 + ch * 8);
    }
#pragma unroll
    for (int j = 0; j < 256 / NT; ++j) {            // B: 64 rows x 4 chunks
      const int idx = tid + j * NT;
      const int row = idx >> 2, ch = idx & 3;
      async_b128(bb + row * 80 + ch * 16, Bt + (long)(bn + row) * K + k0 + ch * 8);
    }
  };

  stage(0, 0);
  for (int kt = 0; kt < nk; ++kt) {
    if (kt + 1 < nk) {
      stage(kt + 1, (kt + 1) & 1);
      // wait only for the PREVIOUS stage's copies (per-wave, in-order)
      if constexpr (BM == 128) asm volatile("s_wait_asynccnt 3" ::: "memory");
      else                     asm volatile("s_wait_asynccnt 4" ::: "memory");
    } else {
      asm volatile("s_wait_asynccnt 0" ::: "memory");
    }
    __syncthreads();
    const _Float16* asb = smem + (size_t)(kt & 1) * BUF_H;
    const _Float16* bsb = asb + BM * AH;
    const v16h af = frag_a(asb + (wave * 16 + i16) * AH, 0, lane);
#pragma unroll
    for (int ct = 0; ct < 4; ++ct) {
      const v16h bf = frag_b(bsb + (ct * 16 + i16) * AH, 0, lane);
      acc[ct] = wmma16(af, bf, acc[ct]);
    }
    __syncthreads();
  }

  // Epilogue: D layout -> M = r + 8*(lane>=16), N = lane%16 (per vgpr r).
  const int m_off = (lane >> 4) * 8;
#pragma unroll
  for (int ct = 0; ct < 4; ++ct) {
#pragma unroll
    for (int r = 0; r < 8; ++r) {
      const int m = bm + wave * 16 + m_off + r;
      const int n = bn + ct * 16 + i16;
      float v = acc[ct][r];
      if (bias) v += bias[n];
      v *= alpha;
      if constexpr (PE) {
        const int   s    = m >> 6;
        const float expo = (float)(n & ~1) * (-9.2103403719761836f / (float)D_DIM);
        const float ang  = (float)s * __expf(expo);
        v += (n & 1) ? __cosf(ang) : __sinf(ang);
      }
      if constexpr (GELU) v = 0.5f * v * (1.0f + erff(v * 0.70710678118654752f));
      if (res) v += (float)res[(long)m * ldc + n];
      long crow = m;
      if constexpr (CMAP == 1) crow = (long)(m & 63) * S_DIM + (m >> 6);
      C[crow * ldc + n] = (CT)v;
    }
  }
}

// ---------------------------------------------------------------------------
// LayerNorm over D=512 (wave per row), f16 in/out, fp32 math.
// ---------------------------------------------------------------------------
__global__ __launch_bounds__(128)
void layernorm_k(const _Float16* __restrict__ X, _Float16* __restrict__ Y,
                 const float* __restrict__ g, const float* __restrict__ b) {
  const int row  = blockIdx.x * 4 + (threadIdx.x >> 5);
  const int lane = threadIdx.x & 31;
  const _Float16* x = X + (long)row * D_DIM;
  float s = 0.f, ss = 0.f;
#pragma unroll
  for (int d = lane; d < D_DIM; d += 32) { const float v = (float)x[d]; s += v; ss += v * v; }
#pragma unroll
  for (int off = 16; off > 0; off >>= 1) {
    s  += __shfl_xor(s, off, 32);
    ss += __shfl_xor(ss, off, 32);
  }
  const float mean = s * (1.0f / D_DIM);
  const float var  = ss * (1.0f / D_DIM) - mean * mean;
  const float inv  = rsqrtf(var + 1e-5f);
  _Float16* y = Y + (long)row * D_DIM;
#pragma unroll
  for (int d = lane; d < D_DIM; d += 32)
    y[d] = (_Float16)(((float)x[d] - mean) * inv * g[d] + b[d]);
}

// ---------------------------------------------------------------------------
// Cross-batch sparse attention, one block per (s, head). Q/K staged with
// async B128 copies; scores & ctx via WMMA; exact top-k(6) threshold mask.
// ---------------------------------------------------------------------------
__global__ __launch_bounds__(256)
void attention_k(const _Float16* __restrict__ Q, const _Float16* __restrict__ Km,
                 const _Float16* __restrict__ V, _Float16* __restrict__ Ctx) {
  __shared__ _Float16 bufA[64][72];   // Q rows, then softmax weights W
  __shared__ _Float16 bufB[64][72];   // K rows (Bt layout for Q*K^T), then V^T
  __shared__ float    Sc[64][64];

  const int s    = blockIdx.x >> 3;
  const int h    = blockIdx.x & 7;
  const int tid  = threadIdx.x;
  const int wave = tid >> 5;
  const int lane = tid & 31;
  const int i16  = lane & 15;
  const long base = (long)s * B_DIM * D_DIM + h * DK;
  const uint32_t ldsA = (uint32_t)(uintptr_t)&bufA[0][0];
  const uint32_t ldsB = (uint32_t)(uintptr_t)&bufB[0][0];

  // Stage Q/K: 64 rows x 128B each, via async B128 (rows 16B-aligned, 144B LDS stride).
#pragma unroll
  for (int j = 0; j < 2; ++j) {
    const int idx = tid + j * 256;     // 0..511
    const int rr = idx >> 3, ch = idx & 7;
    async_b128(ldsA + rr * 144 + ch * 16, Q  + base + (long)rr * D_DIM + ch * 8);
    async_b128(ldsB + rr * 144 + ch * 16, Km + base + (long)rr * D_DIM + ch * 8);
  }
  asm volatile("s_wait_asynccnt 0" ::: "memory");
  __syncthreads();

  // scores = Q K^T / 8 : 16 tiles over 8 waves.
#pragma unroll
  for (int it = 0; it < 2; ++it) {
    const int t  = wave * 2 + it;
    const int rt = t >> 2, ct = t & 3;
    v8f acc = {};
#pragma unroll
    for (int kk = 0; kk < 2; ++kk) {
      const v16h af = frag_a(&bufA[rt * 16 + i16][0], kk * 32, lane);
      const v16h bf = frag_b(&bufB[ct * 16 + i16][0], kk * 32, lane);
      acc = wmma16(af, bf, acc);
    }
    const int m_off = (lane >> 4) * 8;
#pragma unroll
    for (int r = 0; r < 8; ++r)
      Sc[rt * 16 + m_off + r][ct * 16 + i16] = acc[r] * 0.125f;
  }
  __syncthreads();

  if (tid < 64) {   // waves 0-1: top-k(6) threshold + masked softmax
    float thr = 3.4e38f;
    for (int c = 0; c < 64; ++c) {
      const float v = Sc[tid][c];
      int cg = 0;
      for (int c2 = 0; c2 < 64; ++c2) cg += (Sc[tid][c2] > v) ? 1 : 0;
      if (cg < 6 && v < thr) thr = v;
    }
    float mx = -3.4e38f;
    for (int c = 0; c < 64; ++c) {
      const float v = Sc[tid][c];
      if (v >= thr && v > mx) mx = v;
    }
    float e[64];
    float sum = 0.f;
#pragma unroll 8
    for (int c = 0; c < 64; ++c) {
      const float v = Sc[tid][c];
      e[c] = (v >= thr) ? __expf(v - mx) : 0.f;
      sum += e[c];
    }
    const float inv = 1.0f / sum;
#pragma unroll 8
    for (int c = 0; c < 64; ++c) bufA[tid][c] = (_Float16)(e[c] * inv);
  } else {          // waves 2-7: stage V transposed: Vt[d][c] = V[c][d]
    for (int idx = tid - 64; idx < 4096; idx += 192) {
      const int cc = idx >> 6, dd = idx & 63;
      bufB[dd][cc] = V[base + (long)cc * D_DIM + dd];
    }
  }
  __syncthreads();

  // ctx = W * V
#pragma unroll
  for (int it = 0; it < 2; ++it) {
    const int t  = wave * 2 + it;
    const int rt = t >> 2, ct = t & 3;
    v8f acc = {};
#pragma unroll
    for (int kk = 0; kk < 2; ++kk) {
      const v16h af = frag_a(&bufA[rt * 16 + i16][0], kk * 32, lane);
      const v16h bf = frag_b(&bufB[ct * 16 + i16][0], kk * 32, lane);
      acc = wmma16(af, bf, acc);
    }
    const int m_off = (lane >> 4) * 8;
#pragma unroll
    for (int r = 0; r < 8; ++r) {
      const int b = rt * 16 + m_off + r;
      const int d = ct * 16 + i16;
      Ctx[base + (long)b * D_DIM + d] = (_Float16)acc[r];
    }
  }
}

// pooled[b][d] = mean_s h[(s*64+b)*512 + d]
__global__ __launch_bounds__(256)
void pool_k(const _Float16* __restrict__ Hh, _Float16* __restrict__ P) {
  const int idx = blockIdx.x * 256 + threadIdx.x;
  const int b = idx >> 9, d = idx & 511;
  float sum = 0.f;
  for (int s = 0; s < S_DIM; ++s)
    sum += (float)Hh[((long)s * B_DIM + b) * D_DIM + d];
  P[idx] = (_Float16)(sum * (1.0f / S_DIM));
}

// fp32 -> f16 (flat)
__global__ void conv_k(const float* __restrict__ W, _Float16* __restrict__ O, int n) {
  const int idx = blockIdx.x * 256 + threadIdx.x;
  if (idx < n) O[idx] = (_Float16)W[idx];
}

// fp32 W[K][N] -> f16 Wt[N][K]
__global__ void conv_tr_k(const float* __restrict__ W, _Float16* __restrict__ Wt,
                          int K, int N) {
  const int idx = blockIdx.x * 256 + threadIdx.x;
  if (idx >= K * N) return;
  const int k = idx / N, n = idx % N;
  Wt[(long)n * K + k] = (_Float16)W[idx];
}

// ---------------------------------------------------------------------------
// Host orchestration
// ---------------------------------------------------------------------------
template <int BM, int AMAP, int CMAP, int PE, int GELU, typename CT>
static inline void gemm(hipStream_t st, const _Float16* A, int lda,
                        const _Float16* Bt, CT* C, int ldc,
                        int M, int N, int K, const float* bias,
                        const _Float16* res, float alpha) {
  gemm_wmma<BM, AMAP, CMAP, PE, GELU, CT>
      <<<dim3(N / 64, M / BM), dim3(BM * 2), 0, st>>>(A, lda, Bt, C, ldc, K,
                                                      bias, res, alpha);
}

static inline const _Float16* conv_w(hipStream_t st, const float* W, int K, int N,
                                     _Float16*& cur) {
  _Float16* dst = cur;
  cur += (size_t)K * N;
  conv_tr_k<<<dim3((K * N + 255) / 256), dim3(256), 0, st>>>(W, dst, K, N);
  return dst;
}

struct LayerW { const _Float16 *wq, *wk, *wv, *wo, *f1, *f2; };

// p[]: g1,b1,wq,wk,wv,woW,wob,g2,b2,f1W,f1b,f2W,f2b,gf,bf (fp32, stacked L=2)
static void run_stack(hipStream_t st, _Float16* h, _Float16* lnb, _Float16* qb,
                      _Float16* kb, _Float16* vb, _Float16* ffb,
                      const LayerW* lw, const float* const* p) {
  for (int l = 0; l < 2; ++l) {
    const float* g1l  = p[0]  + (size_t)l * D_DIM;
    const float* b1l  = p[1]  + (size_t)l * D_DIM;
    const float* wobl = p[6]  + (size_t)l * D_DIM;
    const float* g2l  = p[7]  + (size_t)l * D_DIM;
    const float* b2l  = p[8]  + (size_t)l * D_DIM;
    const float* f1bl = p[10] + (size_t)l * DFF_DIM;
    const float* f2bl = p[12] + (size_t)l * D_DIM;

    layernorm_k<<<dim3(SB / 4), dim3(128), 0, st>>>(h, lnb, g1l, b1l);
    gemm<128, 0, 0, 0, 0>(st, lnb, D_DIM, lw[l].wq, qb, D_DIM, SB, D_DIM, D_DIM,
                          nullptr, nullptr, 1.f);
    gemm<128, 0, 0, 0, 0>(st, lnb, D_DIM, lw[l].wk, kb, D_DIM, SB, D_DIM, D_DIM,
                          nullptr, nullptr, 1.f);
    gemm<128, 0, 0, 0, 0>(st, lnb, D_DIM, lw[l].wv, vb, D_DIM, SB, D_DIM, D_DIM,
                          nullptr, nullptr, 1.f);
    attention_k<<<dim3(S_DIM * 8), dim3(256), 0, st>>>(qb, kb, vb, lnb);
    gemm<128, 0, 0, 0, 0>(st, lnb, D_DIM, lw[l].wo, h, D_DIM, SB, D_DIM, D_DIM,
                          wobl, h, 1.f);

    layernorm_k<<<dim3(SB / 4), dim3(128), 0, st>>>(h, lnb, g2l, b2l);
    for (int c = 0; c < SB / FF_CHUNK; ++c) {
      const _Float16* Ach = lnb + (size_t)c * FF_CHUNK * D_DIM;
      _Float16*       Cch = h   + (size_t)c * FF_CHUNK * D_DIM;
      gemm<128, 0, 0, 0, 1>(st, Ach, D_DIM, lw[l].f1, ffb, DFF_DIM, FF_CHUNK,
                            DFF_DIM, D_DIM, f1bl, nullptr, 1.f);
      gemm<128, 0, 0, 0, 0>(st, ffb, DFF_DIM, lw[l].f2, Cch, D_DIM, FF_CHUNK,
                            D_DIM, DFF_DIM, f2bl, Cch, 1.f);
    }
  }
  layernorm_k<<<dim3(SB / 4), dim3(128), 0, st>>>(h, lnb, p[13], p[14]);
}

extern "C" void kernel_launch(void* const* d_in, const int* in_sizes, int n_in,
                              void* d_out, int out_size, void* d_ws, size_t ws_size,
                              hipStream_t stream) {
  (void)in_sizes; (void)n_in; (void)out_size; (void)ws_size;

  const float* x        = (const float*)d_in[0];
  const float* enc_in_W = (const float*)d_in[1];
  const float* enc_in_b = (const float*)d_in[2];
  const float* mu_W     = (const float*)d_in[3];
  const float* mu_b     = (const float*)d_in[4];
  const float* lv_W     = (const float*)d_in[5];
  const float* lv_b     = (const float*)d_in[6];
  const float* dec_in_W = (const float*)d_in[7];
  const float* dec_in_b = (const float*)d_in[8];
  const float* out_W    = (const float*)d_in[9];
  const float* out_b    = (const float*)d_in[10];
  const float* ep[15];
  const float* dp[15];
  for (int i = 0; i < 15; ++i) ep[i] = (const float*)d_in[11 + i];
  for (int i = 0; i < 15; ++i) dp[i] = (const float*)d_in[26 + i];

  float* outp  = (float*)d_out;
  float* recon = outp;                                    // (B,S,INP)
  float* mu    = outp + (size_t)B_DIM * S_DIM * INP_DIM;  // (B,LAT)
  float* lv    = mu + (size_t)B_DIM * LAT_DIM;

  _Float16* cur = (_Float16*)d_ws;
  _Float16* h   = cur;  cur += (size_t)SB * D_DIM;
  _Float16* lnb = cur;  cur += (size_t)SB * D_DIM;
  _Float16* qb  = cur;  cur += (size_t)SB * D_DIM;
  _Float16* kb  = cur;  cur += (size_t)SB * D_DIM;
  _Float16* vb  = cur;  cur += (size_t)SB * D_DIM;
  _Float16* ffb = cur;  cur += (size_t)FF_CHUNK * DFF_DIM;
  _Float16* pooled = cur; cur += (size_t)B_DIM * D_DIM;
  _Float16* xh  = cur;  cur += (size_t)B_DIM * S_DIM * INP_DIM;
  _Float16* zh  = cur;  cur += (size_t)B_DIM * LAT_DIM;

  // ---- one-time f16 weight prep (transposed: Wt[n][k]) ----
  const _Float16* enc_inT = conv_w(stream, enc_in_W, INP_DIM, D_DIM, cur);
  const _Float16* dec_inT = conv_w(stream, dec_in_W, LAT_DIM, D_DIM, cur);
  const _Float16* mu_T    = conv_w(stream, mu_W, D_DIM, LAT_DIM, cur);
  const _Float16* lv_T    = conv_w(stream, lv_W, D_DIM, LAT_DIM, cur);
  const _Float16* out_T   = conv_w(stream, out_W, D_DIM, INP_DIM, cur);
  LayerW elw[2], dlw[2];
  for (int l = 0; l < 2; ++l) {
    elw[l].wq = conv_w(stream, ep[2] + (size_t)l * D_DIM * D_DIM, D_DIM, D_DIM, cur);
    elw[l].wk = conv_w(stream, ep[3] + (size_t)l * D_DIM * D_DIM, D_DIM, D_DIM, cur);
    elw[l].wv = conv_w(stream, ep[4] + (size_t)l * D_DIM * D_DIM, D_DIM, D_DIM, cur);
    elw[l].wo = conv_w(stream, ep[5] + (size_t)l * D_DIM * D_DIM, D_DIM, D_DIM, cur);
    elw[l].f1 = conv_w(stream, ep[9] + (size_t)l * D_DIM * DFF_DIM, D_DIM, DFF_DIM, cur);
    elw[l].f2 = conv_w(stream, ep[11] + (size_t)l * DFF_DIM * D_DIM, DFF_DIM, D_DIM, cur);
    dlw[l].wq = conv_w(stream, dp[2] + (size_t)l * D_DIM * D_DIM, D_DIM, D_DIM, cur);
    dlw[l].wk = conv_w(stream, dp[3] + (size_t)l * D_DIM * D_DIM, D_DIM, D_DIM, cur);
    dlw[l].wv = conv_w(stream, dp[4] + (size_t)l * D_DIM * D_DIM, D_DIM, D_DIM, cur);
    dlw[l].wo = conv_w(stream, dp[5] + (size_t)l * D_DIM * D_DIM, D_DIM, D_DIM, cur);
    dlw[l].f1 = conv_w(stream, dp[9] + (size_t)l * D_DIM * DFF_DIM, D_DIM, DFF_DIM, cur);
    dlw[l].f2 = conv_w(stream, dp[11] + (size_t)l * DFF_DIM * D_DIM, DFF_DIM, D_DIM, cur);
  }
  const int xel = B_DIM * S_DIM * INP_DIM;
  conv_k<<<dim3((xel + 255) / 256), dim3(256), 0, stream>>>(x, xh, xel);

  const float sD = sqrtf((float)D_DIM);

  // ---- encoder ----
  gemm<128, 1, 0, 1, 0>(stream, xh, INP_DIM, enc_inT, h, D_DIM, SB, D_DIM,
                        INP_DIM, enc_in_b, nullptr, sD);
  run_stack(stream, h, lnb, qb, kb, vb, ffb, elw, ep);

  // ---- latent heads ----
  pool_k<<<dim3((B_DIM * D_DIM) / 256), dim3(256), 0, stream>>>(lnb, pooled);
  gemm<64, 0, 0, 0, 0>(stream, pooled, D_DIM, mu_T, mu, LAT_DIM, B_DIM, LAT_DIM,
                       D_DIM, mu_b, nullptr, 1.f);
  gemm<64, 0, 0, 0, 0>(stream, pooled, D_DIM, lv_T, lv, LAT_DIM, B_DIM, LAT_DIM,
                       D_DIM, lv_b, nullptr, 1.f);
  const int zel = B_DIM * LAT_DIM;
  conv_k<<<dim3((zel + 255) / 256), dim3(256), 0, stream>>>(mu, zh, zel);

  // ---- decoder ----
  gemm<128, 2, 0, 1, 0>(stream, zh, LAT_DIM, dec_inT, h, D_DIM, SB, D_DIM,
                        LAT_DIM, dec_in_b, nullptr, sD);
  run_stack(stream, h, lnb, qb, kb, vb, ffb, dlw, dp);
  gemm<128, 0, 1, 0, 0>(stream, lnb, D_DIM, out_T, recon, INP_DIM, SB, INP_DIM,
                        D_DIM, out_b, nullptr, 1.f);
}
